// WeatherMambaSegmentationFinal_29695403885131
// MI455X (gfx1250) — matrix operand order repaired and checked by reading, not compiled, not run on hardware
//
#include <hip/hip_runtime.h>
#include <hip/hip_bf16.h>
#include <stdint.h>
#include <stddef.h>

// ---------------------------------------------------------------------------
// Model constants
// ---------------------------------------------------------------------------
#define BB     2
#define LL     2048
#define DD     384
#define DI     768
#define NST    16
#define MROWS  (BB * LL)     // 4096
#define DTRANK 24

#define ACT_NONE     0
#define ACT_GELU     1
#define ACT_SILU     2
#define ACT_SOFTPLUS 3
#define ACT_SIGMOID  4
#define ACT_RELU     5

typedef __bf16 bf16_t;
typedef __attribute__((ext_vector_type(8)))  bf16_t bf16x8;
typedef __attribute__((ext_vector_type(16))) bf16_t bf16x16;
typedef __attribute__((ext_vector_type(8)))  float  f32x8;

static inline int cdiv(int a, int b) { return (a + b - 1) / b; }

// ---------------------------------------------------------------------------
// Device helpers
// ---------------------------------------------------------------------------
__device__ __forceinline__ float actf(float v, int act) {
  switch (act) {
    case ACT_GELU:     return 0.5f * v * (1.f + erff(v * 0.70710678118654752f));
    case ACT_SILU:     return v / (1.f + __expf(-v));
    case ACT_SOFTPLUS: return (v > 20.f) ? v : log1pf(__expf(v));
    case ACT_SIGMOID:  return 1.f / (1.f + __expf(-v));
    case ACT_RELU:     return v > 0.f ? v : 0.f;
    default:           return v;
  }
}

// CDNA5 async copy: global -> LDS, 16 bytes, tracked by ASYNCcnt.
__device__ __forceinline__ void async_ld_b128(uint32_t lds_off, const void* g) {
  unsigned long long ga = (unsigned long long)g;
  asm volatile("global_load_async_to_lds_b128 %0, %1, off"
               :: "v"(lds_off), "v"(ga) : "memory");
}
// Wait until this wave's ASYNCcnt <= n (n is block-uniform, 0/2/4).
__device__ __forceinline__ void wait_async_le(int n) {
  switch (n) {
    case 0:  asm volatile("s_wait_asynccnt 0" ::: "memory"); break;
    case 2:  asm volatile("s_wait_asynccnt 2" ::: "memory"); break;
    default: asm volatile("s_wait_asynccnt 4" ::: "memory"); break;
  }
}

// ---------------------------------------------------------------------------
// fp32 -> bf16 strided conversion (also used to build concat inputs)
// ---------------------------------------------------------------------------
__global__ void cvt_kernel(const float* __restrict__ in, bf16_t* __restrict__ out,
                           int rows, int cols, int ldi, int ldo) {
  int i = blockIdx.x * blockDim.x + threadIdx.x;
  if (i >= rows * cols) return;
  int r = i / cols, c = i - r * cols;
  out[(size_t)r * ldo + c] = (bf16_t)in[(size_t)r * ldi + c];
}

// ---------------------------------------------------------------------------
// WMMA GEMM:  C[M,N] = act( A[M,K](bf16) * W[N,K]^T(bf16) + bias[N] )
// 64x64x32 tile / 128-thread block (4 waves).
// Double-buffered LDS staged by global_load_async_to_lds_b128; row pad to
// 40 bf16 (80 B = 20 banks, 16B-aligned) to avoid bank conflicts.
// ---------------------------------------------------------------------------
#define LPAD 40
__global__ __launch_bounds__(128)
void gemm_kernel(const bf16_t* __restrict__ A, int lda,
                 const bf16_t* __restrict__ W, int ldw,
                 const float* __restrict__ bias,
                 float* __restrict__ C, int ldc,
                 int M, int N, int K, int act) {
  __shared__ __align__(16) bf16_t As[2][64][LPAD];
  __shared__ __align__(16) bf16_t Bs[2][64][LPAD];

  const int tid  = threadIdx.x;
  const int wave = tid >> 5;
  const int lane = tid & 31;
  const int m0 = blockIdx.y * 64;
  const int n0 = blockIdx.x * 64;
  const int r  = tid >> 1;        // 0..63 : tile row
  const int h  = tid & 1;         // 0/1   : 16-element K half

  // Block-uniform fast-path predicates -> every wave issues the same number
  // of async instructions per tile (required for the asynccnt pipeline).
  const bool okM = (m0 + 64 <= M) && ((lda & 7) == 0);
  const bool okN = (n0 + 64 <= N) && ((ldw & 7) == 0);

  f32x8 acc[4];
#pragma unroll
  for (int t = 0; t < 4; ++t)
#pragma unroll
    for (int e = 0; e < 8; ++e) acc[t][e] = 0.f;

  // Stage one 64x32 K-slice of A and W into LDS buffer `buf`.
  // Returns #async instructions issued (block-uniform: 0/2/4).
  auto stage = [&](int k0, int buf) -> int {
    int cnt = 0;
    const int gk = k0 + h * 16;
    const bool kfull = (k0 + 32 <= K);
    // ---- A tile ----
    if (okM && kfull) {
      uint32_t lo = (uint32_t)(uintptr_t)&As[buf][r][h * 16];
      const bf16_t* gp = A + (size_t)(m0 + r) * lda + gk;
      async_ld_b128(lo, gp);
      async_ld_b128(lo + 16, gp + 8);
      if (k0 + 64 < K) __builtin_prefetch(gp + 64, 0, 1);
      cnt += 2;
    } else {
#pragma unroll
      for (int i = 0; i < 16; ++i) {
        int kk = gk + i;
        bf16_t v = (bf16_t)0.f;
        if ((m0 + r) < M && kk < K) v = A[(size_t)(m0 + r) * lda + kk];
        As[buf][r][h * 16 + i] = v;
      }
    }
    // ---- W tile (rows = output columns, K contiguous) ----
    if (okN && kfull) {
      uint32_t lo = (uint32_t)(uintptr_t)&Bs[buf][r][h * 16];
      const bf16_t* gp = W + (size_t)(n0 + r) * ldw + gk;
      async_ld_b128(lo, gp);
      async_ld_b128(lo + 16, gp + 8);
      cnt += 2;
    } else {
#pragma unroll
      for (int i = 0; i < 16; ++i) {
        int kk = gk + i;
        bf16_t v = (bf16_t)0.f;
        if ((n0 + r) < N && kk < K) v = W[(size_t)(n0 + r) * ldw + kk];
        Bs[buf][r][h * 16 + i] = v;
      }
    }
    return cnt;
  };

  const int nk = (K + 31) / 32;
  (void)stage(0, 0);                         // prologue: tile 0 -> buf 0
  for (int i = 0; i < nk; ++i) {
    int nxt = 0;
    if (i + 1 < nk) nxt = stage((i + 1) * 32, (i + 1) & 1);   // overlap copy
    wait_async_le(nxt);                      // tile i complete (this wave)
    __syncthreads();                         // ... for all waves

    const int buf  = i & 1;
    const int arow = wave * 16 + (lane & 15);
    const int hi   = lane >> 4;
    // A fragment: lanes 0-15 hold K 0..7 & 16..23, lanes 16-31 hold 8..15 & 24..31
    union { bf16x8 hh[2]; bf16x16 v; } ua;
    ua.hh[0] = *(const bf16x8*)&As[buf][arow][hi * 8];
    ua.hh[1] = *(const bf16x8*)&As[buf][arow][16 + hi * 8];
    const bf16x16 afrag = ua.v;
    // Hoist all B fragments, then issue 4 back-to-back WMMAs.
    bf16x16 bfrag[4];
#pragma unroll
    for (int t = 0; t < 4; ++t) {
      union { bf16x8 hh[2]; bf16x16 v; } ub;
      ub.hh[0] = *(const bf16x8*)&Bs[buf][t * 16 + (lane & 15)][hi * 8];
      ub.hh[1] = *(const bf16x8*)&Bs[buf][t * 16 + (lane & 15)][16 + hi * 8];
      bfrag[t] = ub.v;
    }
#pragma unroll
    for (int t = 0; t < 4; ++t)
      acc[t] = __builtin_amdgcn_wmma_f32_16x16x32_bf16(
                   false, afrag, false, bfrag[t], (short)0, acc[t], false, false);
    __syncthreads();                         // buf reusable for tile i+2
  }

  // C/D layout: VGPR e -> row e (+8 for lanes 16-31), col = lane&15.
#pragma unroll
  for (int t = 0; t < 4; ++t) {
    const int col = n0 + t * 16 + (lane & 15);
    if (col >= N) continue;
    const float bv = bias ? bias[col] : 0.f;
#pragma unroll
    for (int e = 0; e < 8; ++e) {
      const int row = m0 + wave * 16 + (lane >> 4) * 8 + e;
      if (row < M) C[(size_t)row * ldc + col] = actf(acc[t][e] + bv, act);
    }
  }
}

// ---------------------------------------------------------------------------
// LayerNorm over last dim (one row per block, 128 threads) + optional act
// ---------------------------------------------------------------------------
__global__ __launch_bounds__(128)
void ln_kernel(const float* __restrict__ x, int ldx,
               const float* __restrict__ g, const float* __restrict__ b,
               float* __restrict__ out, int ldo, int Cc, int act) {
  const size_t row = blockIdx.x;
  const float* xr = x + row * ldx;
  __shared__ float red[128];
  const int tid = threadIdx.x;

  float s = 0.f;
  for (int c = tid; c < Cc; c += 128) s += xr[c];
  red[tid] = s; __syncthreads();
  for (int o = 64; o > 0; o >>= 1) { if (tid < o) red[tid] += red[tid + o]; __syncthreads(); }
  const float mean = red[0] / Cc; __syncthreads();

  float v = 0.f;
  for (int c = tid; c < Cc; c += 128) { float d = xr[c] - mean; v += d * d; }
  red[tid] = v; __syncthreads();
  for (int o = 64; o > 0; o >>= 1) { if (tid < o) red[tid] += red[tid + o]; __syncthreads(); }
  const float rstd = rsqrtf(red[0] / Cc + 1e-5f);

  for (int c = tid; c < Cc; c += 128) {
    float t = (xr[c] - mean) * rstd * g[c] + b[c];
    out[row * ldo + c] = actf(t, act);
  }
}

// ---------------------------------------------------------------------------
// Depthwise conv along L (zero pad), torch weight layout w[c*Kc + k]
// ---------------------------------------------------------------------------
__global__ void dwconv_kernel(const float* __restrict__ x, int ldx,
                              const float* __restrict__ w, const float* __restrict__ bias,
                              float* __restrict__ out, int ldo,
                              int L, int Cc, int Kc, int padl, int act) {
  int idx = blockIdx.x * blockDim.x + threadIdx.x;
  int total = BB * L * Cc;
  if (idx >= total) return;
  int c = idx % Cc;
  int t = (idx / Cc) % L;
  int b = idx / (Cc * L);
  float s = bias[c];
  for (int k = 0; k < Kc; ++k) {
    int pos = t + k - padl;
    if (pos >= 0 && pos < L) s += x[((size_t)b * L + pos) * ldx + c] * w[c * Kc + k];
  }
  out[((size_t)b * L + t) * ldo + c] = actf(s, act);
}

// ---------------------------------------------------------------------------
// Mamba selective scan: one channel per thread, h[16] in VGPRs.
// B_t/C_t broadcast via double-buffered LDS: one barrier per step; next
// step's 32 loads overlap the current step's exp/fma chain.
// ---------------------------------------------------------------------------
__global__ __launch_bounds__(256)
void scan_kernel(const float* __restrict__ dt,
                 const float* __restrict__ Bm, int ldbc,
                 const float* __restrict__ Cm,
                 const float* __restrict__ u, int ldu,
                 const float* __restrict__ A_log, const float* __restrict__ Dp,
                 float* __restrict__ y, int L, int di) {
  const int b = blockIdx.y;
  const int d = blockIdx.x * 256 + threadIdx.x;   // di == gridDim.x*256
  const int tid = threadIdx.x;
  float a[NST], h[NST];
#pragma unroll
  for (int n = 0; n < NST; ++n) { a[n] = -__expf(A_log[(size_t)d * NST + n]); h[n] = 0.f; }
  const float Dv = Dp[d];
  __shared__ float sB[2][NST], sC[2][NST];

  const size_t base = (size_t)b * L;
  if (tid < NST)          sB[0][tid]       = Bm[base * ldbc + tid];
  else if (tid < 2 * NST) sC[0][tid - NST] = Cm[base * ldbc + tid - NST];
  __syncthreads();

  for (int t = 0; t < L; ++t) {
    const size_t r = base + t;
    const int cur = t & 1, nx = cur ^ 1;
    if (t + 1 < L) {
      if (tid < NST)          sB[nx][tid]       = Bm[(r + 1) * ldbc + tid];
      else if (tid < 2 * NST) sC[nx][tid - NST] = Cm[(r + 1) * ldbc + tid - NST];
    }
    const float dtv = dt[r * di + d];
    const float uv  = u[r * ldu + d];
    const float du  = dtv * uv;
    float acc = 0.f;
#pragma unroll
    for (int n = 0; n < NST; ++n) {
      h[n] = __expf(dtv * a[n]) * h[n] + du * sB[cur][n];
      acc += h[n] * sC[cur][n];
    }
    y[r * di + d] = acc + uv * Dv;
    __syncthreads();
  }
}

// ---------------------------------------------------------------------------
// Elementwise kernels
// ---------------------------------------------------------------------------
__global__ void gate_kernel(const float* __restrict__ y, const float* __restrict__ z,
                            int ldz, float* __restrict__ out, int rows, int cols) {
  int i = blockIdx.x * blockDim.x + threadIdx.x;
  if (i >= rows * cols) return;
  int r = i / cols, c = i - r * cols;
  float zv = z[(size_t)r * ldz + c];
  out[(size_t)r * cols + c] = y[(size_t)r * cols + c] * (zv / (1.f + __expf(-zv)));
}

__global__ void add_kernel(const float* __restrict__ a, const float* __restrict__ b,
                           float* __restrict__ out, int n) {
  int i = blockIdx.x * blockDim.x + threadIdx.x;
  if (i < n) out[i] = a[i] + b[i];
}

__global__ void flip_kernel(const float* __restrict__ in, float* __restrict__ out,
                            int L, int Cc) {
  int i = blockIdx.x * blockDim.x + threadIdx.x;
  if (i >= BB * L * Cc) return;
  int c = i % Cc;
  int l = (i / Cc) % L;
  int b = i / (Cc * L);
  out[((size_t)b * L + l) * Cc + c] = in[((size_t)b * L + (L - 1 - l)) * Cc + c];
}

// MANF: edge = mean(neighbors, replicate-pad, offset 0 excluded) - x
__global__ void edge_kernel(const float* __restrict__ x, float* __restrict__ out,
                            int L, int Cc, int p) {
  int i = blockIdx.x * blockDim.x + threadIdx.x;
  if (i >= BB * L * Cc) return;
  int c = i % Cc;
  int l = (i / Cc) % L;
  int b = i / (Cc * L);
  float s = 0.f;
  for (int o = -p; o <= p; ++o) {
    if (o == 0) continue;
    int ln = l + o; ln = ln < 0 ? 0 : (ln >= L ? L - 1 : ln);
    s += x[((size_t)b * L + ln) * Cc + c];
  }
  out[i] = s / (2.f * p) - x[((size_t)b * L + l) * Cc + c];
}

// Geometry features: [coords(3), mean(3), std(3), mean-dist(1)] over 16 neighbors
__global__ void geo_kernel(const float* __restrict__ coords, float* __restrict__ gf, int L) {
  int row = blockIdx.x * blockDim.x + threadIdx.x;
  if (row >= BB * L) return;
  int b = row / L, l = row - b * L;
  float cx[3], m[3] = {0, 0, 0}, s2[3] = {0, 0, 0};
  for (int d = 0; d < 3; ++d) cx[d] = coords[(size_t)row * 3 + d];
  float gd = 0.f;
  for (int j = 0; j < 16; ++j) {
    int o = (j < 8) ? (j - 8) : (j - 7);
    int ln = l + o; ln = ln < 0 ? 0 : (ln >= L ? L - 1 : ln);
    size_t nr = (size_t)b * L + ln;
    float dd = 0.f;
    for (int d = 0; d < 3; ++d) {
      float v = coords[nr * 3 + d];
      m[d] += v; s2[d] += v * v;
      float df = v - cx[d]; dd += df * df;
    }
    gd += sqrtf(dd);
  }
  for (int d = 0; d < 3; ++d) gf[(size_t)row * 10 + d] = cx[d];
  for (int d = 0; d < 3; ++d) {
    float mm = m[d] / 16.f;
    gf[(size_t)row * 10 + 3 + d] = mm;
    float var = s2[d] / 16.f - mm * mm;
    gf[(size_t)row * 10 + 6 + d] = sqrtf(var > 0.f ? var : 0.f);
  }
  gf[(size_t)row * 10 + 9] = gd / 16.f;
}

// RADM: per-channel neighbor mean/std over 16 offsets
__global__ void meanstd_kernel(const float* __restrict__ x, float* __restrict__ lmf,
                               float* __restrict__ lsf, int L, int Cc) {
  int i = blockIdx.x * blockDim.x + threadIdx.x;
  if (i >= BB * L * Cc) return;
  int c = i % Cc;
  int l = (i / Cc) % L;
  int b = i / (Cc * L);
  float s = 0.f, s2 = 0.f;
  for (int j = 0; j < 16; ++j) {
    int o = (j < 8) ? (j - 8) : (j - 7);
    int ln = l + o; ln = ln < 0 ? 0 : (ln >= L ? L - 1 : ln);
    float v = x[((size_t)b * L + ln) * Cc + c];
    s += v; s2 += v * v;
  }
  float mm = s / 16.f;
  lmf[i] = mm;
  float var = s2 / 16.f - mm * mm;
  lsf[i] = sqrtf(var > 0.f ? var : 0.f);
}

// RADM per-row scalars: [lm, ls, dens, sv]
__global__ __launch_bounds__(128)
void rowstat_kernel(const float* __restrict__ x, const float* __restrict__ lmf,
                    const float* __restrict__ lsf, float* __restrict__ out4,
                    int L, int Cc) {
  const int row = blockIdx.x;
  const int b = row / L, l = row - b * L;
  const int tid = threadIdx.x;
  __shared__ float red[128];
  __shared__ float sres[2];

  float a0 = 0.f, a1 = 0.f;
  for (int c = tid; c < Cc; c += 128) {
    a0 += lmf[(size_t)row * Cc + c];
    a1 += lsf[(size_t)row * Cc + c];
  }
  red[tid] = a0; __syncthreads();
  for (int o = 64; o > 0; o >>= 1) { if (tid < o) red[tid] += red[tid + o]; __syncthreads(); }
  if (tid == 0) sres[0] = red[0] / Cc;
  __syncthreads();
  red[tid] = a1; __syncthreads();
  for (int o = 64; o > 0; o >>= 1) { if (tid < o) red[tid] += red[tid + o]; __syncthreads(); }
  if (tid == 0) sres[1] = red[0] / Cc;
  __syncthreads();

  float dens = 0.f, sv = 0.f;
  for (int j = 0; j < 16; ++j) {
    int o = (j < 8) ? (j - 8) : (j - 7);
    int ln = l + o; ln = ln < 0 ? 0 : (ln >= L ? L - 1 : ln);
    size_t nr = (size_t)b * L + ln;
    float loc = 0.f;
    for (int c = tid; c < Cc; c += 128) {
      float d = x[nr * Cc + c] - x[(size_t)row * Cc + c];
      loc += d * d;
    }
    red[tid] = loc; __syncthreads();
    for (int oo = 64; oo > 0; oo >>= 1) { if (tid < oo) red[tid] += red[tid + oo]; __syncthreads(); }
    if (tid == 0) { dens += sqrtf(red[0]); sv += red[0]; }
    __syncthreads();
  }
  if (tid == 0) {
    out4[(size_t)row * 4 + 0] = sres[0];
    out4[(size_t)row * 4 + 1] = sres[1];
    out4[(size_t)row * 4 + 2] = dens / 16.f;
    out4[(size_t)row * 4 + 3] = sv / (16.f * Cc);
  }
}

__global__ void fscd_kernel(const float* __restrict__ f0, const float* __restrict__ f1,
                            const float* __restrict__ f2, float* __restrict__ fs,
                            float* __restrict__ cd, int n) {
  int i = blockIdx.x * blockDim.x + threadIdx.x;
  if (i >= n) return;
  fs[i] = f0[i] + f1[i] + f2[i];
  cd[i] = f2[i] - f0[i];
}

// out = (1-p)*x + p*y, p broadcast per row
__global__ void blend_kernel(const float* __restrict__ x, const float* __restrict__ y,
                             const float* __restrict__ p, float* __restrict__ out,
                             int rows, int cols) {
  int i = blockIdx.x * blockDim.x + threadIdx.x;
  if (i >= rows * cols) return;
  int r = i / cols;
  float pv = p[r];
  out[i] = (1.f - pv) * x[i] + pv * y[i];
}

// ---------------------------------------------------------------------------
// Host-side parameter views (JAX pytree leaf order: dict keys sorted)
// ---------------------------------------------------------------------------
struct Lin  { const float* b; const float* w; };
struct LNp  { const float* b; const float* g; };
struct Mlp3 { Lin l1, l2; LNp ln; };
struct MambaP {
  const float *A_log, *D, *conv_b, *conv_w, *dt_b, *dt_w, *in_w, *out_w, *x_w;
};
struct BlockP { MambaP bwd; Lin fuse_lin; LNp fuse_ln; MambaP fwd; LNp pre; };
struct StageP { BlockP blk[2]; Lin in_lin; LNp in_ln; const float* mix_b; const float* mix_w; LNp norm; };

struct Env {
  hipStream_t s;
  float *t0, *t1, *t2, *t3, *xA, *xB;
  float *xz, *xm, *dbl, *dtb, *ycore;
  float *lmf, *lsf, *sc4, *sc1a, *sc1b, *gf, *g96;
  bf16_t *abf, *abf2, *wbf;
};

static void cvtf(Env& e, const float* in, int ldi, bf16_t* out, int ldo, int rows, int cols) {
  cvt_kernel<<<cdiv(rows * cols, 256), 256, 0, e.s>>>(in, out, rows, cols, ldi, ldo);
}
static void gemmf(Env& e, const bf16_t* A, int lda, const float* Wf, int N, int K,
                  const float* bias, float* C, int ldc, int M, int act) {
  cvt_kernel<<<cdiv(N * K, 256), 256, 0, e.s>>>(Wf, e.wbf, N, K, K, K);
  dim3 g(cdiv(N, 64), cdiv(M, 64));
  gemm_kernel<<<g, 128, 0, e.s>>>(A, lda, e.wbf, K, bias, C, ldc, M, N, K, act);
}
static void lnf(Env& e, const float* x, int ldx, const LNp& p, float* out, int ldo,
                int M, int Cc, int act) {
  ln_kernel<<<M, 128, 0, e.s>>>(x, ldx, p.g, p.b, out, ldo, Cc, act);
}
static void addf(Env& e, const float* a, const float* b, float* out, int n) {
  add_kernel<<<cdiv(n, 256), 256, 0, e.s>>>(a, b, out, n);
}
static void flipf(Env& e, const float* in, float* out) {
  flip_kernel<<<cdiv(MROWS * DD, 256), 256, 0, e.s>>>(in, out, LL, DD);
}

// mlp3: Linear -> LN -> GELU -> Linear (+act on final)
static void run_mlp3(Env& e, const Mlp3& m, const bf16_t* inbf, int ldin, int Kin,
                     int Hid, int Nout, float* out, int ldo, int act_out) {
  gemmf(e, inbf, ldin, m.l1.w, Hid, Kin, m.l1.b, e.dtb, Hid, MROWS, ACT_NONE);
  lnf(e, e.dtb, Hid, m.ln, e.dtb, Hid, MROWS, Hid, ACT_GELU);
  cvtf(e, e.dtb, Hid, e.abf2, Hid, MROWS, Hid);
  gemmf(e, e.abf2, Hid, m.l2.w, Nout, Hid, m.l2.b, out, ldo, MROWS, act_out);
}

// Full Mamba branch: in (M,384) fp32 -> out (M,384) fp32
static void run_mamba(Env& e, const MambaP& mp, const float* xin, float* out) {
  cvtf(e, xin, DD, e.abf, DD, MROWS, DD);
  gemmf(e, e.abf, DD, mp.in_w, 2 * DI, DD, nullptr, e.xz, 2 * DI, MROWS, ACT_NONE);
  // causal depthwise conv (K=4, pad 3/0) + SiLU on xm half of xz
  dwconv_kernel<<<cdiv(MROWS * DI, 256), 256, 0, e.s>>>(
      e.xz, 2 * DI, mp.conv_w, mp.conv_b, e.xm, DI, LL, DI, 4, 3, ACT_SILU);
  cvtf(e, e.xm, DI, e.abf2, DI, MROWS, DI);
  gemmf(e, e.abf2, DI, mp.x_w, DTRANK + 2 * NST, DI, nullptr, e.dbl, 56, MROWS, ACT_NONE);
  // dt = softplus(dt_raw @ dt_w^T + dt_b)
  cvtf(e, e.dbl, 56, e.abf, DTRANK, MROWS, DTRANK);
  gemmf(e, e.abf, DTRANK, mp.dt_w, DI, DTRANK, mp.dt_b, e.dtb, DI, MROWS, ACT_SOFTPLUS);
  // selective scan (folds + u*D)
  scan_kernel<<<dim3(DI / 256, BB), 256, 0, e.s>>>(
      e.dtb, e.dbl + DTRANK, 56, e.dbl + DTRANK + NST, e.xm, DI,
      mp.A_log, mp.D, e.ycore, LL, DI);
  // gate by silu(z) in-place
  gate_kernel<<<cdiv(MROWS * DI, 256), 256, 0, e.s>>>(
      e.ycore, e.xz + DI, 2 * DI, e.ycore, MROWS, DI);
  cvtf(e, e.ycore, DI, e.abf2, DI, MROWS, DI);
  gemmf(e, e.abf2, DI, mp.out_w, DD, DI, nullptr, out, DD, MROWS, ACT_NONE);
}

// ---------------------------------------------------------------------------
// kernel_launch
// ---------------------------------------------------------------------------
extern "C" void kernel_launch(void* const* d_in, const int* in_sizes, int n_in,
                              void* d_out, int out_size, void* d_ws, size_t ws_size,
                              hipStream_t stream) {
  (void)in_sizes; (void)n_in; (void)out_size; (void)ws_size;
  const float* x_in   = (const float*)d_in[0];
  const float* coords = (const float*)d_in[1];
  float* out = (float*)d_out;

  // ---- walk params (pytree: dict keys sorted alphabetically) ----
  int ci = 2;
  auto nf = [&]() { return (const float*)d_in[ci++]; };
  auto rdLin = [&](Lin& x)   { x.b = nf(); x.w = nf(); };
  auto rdLN  = [&](LNp& x)   { x.b = nf(); x.g = nf(); };
  auto rdM3  = [&](Mlp3& m)  { rdLin(m.l1); rdLin(m.l2); rdLN(m.ln); };
  auto rdMb  = [&](MambaP& m){ m.A_log = nf(); m.D = nf(); m.conv_b = nf(); m.conv_w = nf();
                               m.dt_b = nf(); m.dt_w = nf(); m.in_w = nf();
                               m.out_w = nf(); m.x_w = nf(); };

  LNp final_ln; rdLN(final_ln);
  Lin head;     rdLin(head);
  Mlp3 fus;     rdM3(fus);
  Lin geo_l1, geo_l2; rdLin(geo_l1); rdLin(geo_l2);
  Mlp3 large;   rdM3(large);
  Mlp3 medium;  rdM3(medium);
  LNp manf_norm; rdLN(manf_norm);
  Mlp3 small;   rdM3(small);
  Mlp3 cn;      rdM3(cn);
  Mlp3 nd;      rdM3(nd);
  LNp radm_norm; rdLN(radm_norm);
  Mlp3 rb;      rdM3(rb);
  StageP st[3];
  for (int i = 0; i < 3; ++i) {
    for (int b = 0; b < 2; ++b) {
      rdMb(st[i].blk[b].bwd); rdLin(st[i].blk[b].fuse_lin);
      rdLN(st[i].blk[b].fuse_ln); rdMb(st[i].blk[b].fwd); rdLN(st[i].blk[b].pre);
    }
    rdLin(st[i].in_lin); rdLN(st[i].in_ln);
    st[i].mix_b = nf(); st[i].mix_w = nf(); rdLN(st[i].norm);
  }
  Lin tr_lin[2]; LNp tr_ln[2];
  for (int i = 0; i < 2; ++i) { rdLin(tr_lin[i]); rdLN(tr_ln[i]); }

  // ---- workspace carve ----
  uint8_t* wp = (uint8_t*)d_ws;
  auto carve = [&](size_t bytes) {
    void* p = wp; wp += (bytes + 255) & ~(size_t)255; return p;
  };
  Env e; e.s = stream;
  e.xA    = (float*)carve(sizeof(float) * MROWS * DD);
  e.xB    = (float*)carve(sizeof(float) * MROWS * DD);
  e.t0    = (float*)carve(sizeof(float) * MROWS * DD);
  e.t1    = (float*)carve(sizeof(float) * MROWS * DD);
  e.t2    = (float*)carve(sizeof(float) * MROWS * DD);
  e.t3    = (float*)carve(sizeof(float) * MROWS * DD);
  e.xz    = (float*)carve(sizeof(float) * MROWS * 2 * DI);
  e.xm    = (float*)carve(sizeof(float) * MROWS * DI);
  e.dtb   = (float*)carve(sizeof(float) * MROWS * DI);
  e.ycore = (float*)carve(sizeof(float) * MROWS * DI);
  e.dbl   = (float*)carve(sizeof(float) * MROWS * 56);
  e.lmf   = (float*)carve(sizeof(float) * MROWS * DD);
  e.lsf   = (float*)carve(sizeof(float) * MROWS * DD);
  e.sc4   = (float*)carve(sizeof(float) * MROWS * 4);
  e.sc1a  = (float*)carve(sizeof(float) * MROWS);
  e.sc1b  = (float*)carve(sizeof(float) * MROWS);
  e.gf    = (float*)carve(sizeof(float) * MROWS * 10);
  e.g96   = (float*)carve(sizeof(float) * MROWS * 96);
  e.abf   = (bf16_t*)carve(sizeof(bf16_t) * MROWS * 4 * DD);
  e.abf2  = (bf16_t*)carve(sizeof(bf16_t) * MROWS * DI);
  e.wbf   = (bf16_t*)carve(sizeof(bf16_t) * 1536 * 384);

  // ================= backbone =================
  const float* src = x_in;
  float* xcur = e.xA;
  for (int si = 0; si < 3; ++si) {
    StageP& S = st[si];
    cvtf(e, src, DD, e.abf, DD, MROWS, DD);
    gemmf(e, e.abf, DD, S.in_lin.w, DD, DD, S.in_lin.b, e.t0, DD, MROWS, ACT_NONE);
    lnf(e, e.t0, DD, S.in_ln, xcur, DD, MROWS, DD, ACT_GELU);
    for (int bi = 0; bi < 2; ++bi) {
      BlockP& Bk = S.blk[bi];
      lnf(e, xcur, DD, Bk.pre, e.t0, DD, MROWS, DD, ACT_NONE);    // xn
      run_mamba(e, Bk.fwd, e.t0, e.t1);                           // xf
      flipf(e, e.t0, e.t2);
      run_mamba(e, Bk.bwd, e.t2, e.t3);
      flipf(e, e.t3, e.t2);                                       // xb
      cvtf(e, e.t1, DD, e.abf, 2 * DD, MROWS, DD);
      cvtf(e, e.t2, DD, e.abf + DD, 2 * DD, MROWS, DD);
      gemmf(e, e.abf, 2 * DD, Bk.fuse_lin.w, DD, 2 * DD, Bk.fuse_lin.b,
            e.t3, DD, MROWS, ACT_NONE);
      lnf(e, e.t3, DD, Bk.fuse_ln, e.t3, DD, MROWS, DD, ACT_GELU);
      addf(e, xcur, e.t3, xcur, MROWS * DD);
    }
    dwconv_kernel<<<cdiv(MROWS * DD, 256), 256, 0, stream>>>(
        xcur, DD, S.mix_w, S.mix_b, e.t0, DD, LL, DD, 3, 1, ACT_NONE);
    addf(e, xcur, e.t0, e.t0, MROWS * DD);
    lnf(e, e.t0, DD, S.norm, xcur, DD, MROWS, DD, ACT_NONE);
    if (si < 2) {
      cvtf(e, xcur, DD, e.abf, DD, MROWS, DD);
      gemmf(e, e.abf, DD, tr_lin[si].w, DD, DD, tr_lin[si].b, e.t0, DD, MROWS, ACT_NONE);
      lnf(e, e.t0, DD, tr_ln[si], xcur, DD, MROWS, DD, ACT_GELU);
    }
    src = xcur;
  }
  lnf(e, xcur, DD, final_ln, e.xB, DD, MROWS, DD, ACT_NONE);
  float* X = e.xB;

  // ================= MANF =================
  float* feat[3] = {e.t1, e.t2, e.t3};
  const Mlp3* fm[3] = {&small, &medium, &large};
  const int ps[3] = {4, 8, 16};
  for (int i = 0; i < 3; ++i) {
    edge_kernel<<<cdiv(MROWS * DD, 256), 256, 0, stream>>>(X, e.t0, LL, DD, ps[i]);
    cvtf(e, X, DD, e.abf, 2 * DD, MROWS, DD);
    cvtf(e, e.t0, DD, e.abf + DD, 2 * DD, MROWS, DD);
    run_mlp3(e, *fm[i], e.abf, 2 * DD, 2 * DD, DD, DD, feat[i], DD, ACT_NONE);
  }
  geo_kernel<<<cdiv(MROWS, 256), 256, 0, stream>>>(coords, e.gf, LL);
  cvtf(e, e.gf, 10, e.abf, 10, MROWS, 10);
  gemmf(e, e.abf, 10, geo_l1.w, 96, 10, geo_l1.b, e.g96, 96, MROWS, ACT_RELU);
  cvtf(e, e.g96, 96, e.abf2, 96, MROWS, 96);
  gemmf(e, e.abf2, 96, geo_l2.w, DD, 96, geo_l2.b, e.t0, DD, MROWS, ACT_NONE);  // geo
  fscd_kernel<<<cdiv(MROWS * DD, 256), 256, 0, stream>>>(
      e.t1, e.t2, e.t3, e.xm, e.ycore, MROWS * DD);   // fs -> xm, cd -> ycore
  cvtf(e, X, DD, e.abf, 4 * DD, MROWS, DD);
  cvtf(e, e.xm, DD, e.abf + DD, 4 * DD, MROWS, DD);
  cvtf(e, e.t0, DD, e.abf + 2 * DD, 4 * DD, MROWS, DD);
  cvtf(e, e.ycore, DD, e.abf + 3 * DD, 4 * DD, MROWS, DD);
  run_mlp3(e, fus, e.abf, 4 * DD, 4 * DD, DD, DD, e.t1, DD, ACT_NONE);
  addf(e, X, e.t1, e.t1, MROWS * DD);
  lnf(e, e.t1, DD, manf_norm, e.xA, DD, MROWS, DD, ACT_NONE);
  X = e.xA;

  // ================= RADM =================
  meanstd_kernel<<<cdiv(MROWS * DD, 256), 256, 0, stream>>>(X, e.lmf, e.lsf, LL, DD);
  rowstat_kernel<<<MROWS, 128, 0, stream>>>(X, e.lmf, e.lsf, e.sc4, LL, DD);
  cvtf(e, X, DD, e.abf, DD + 4, MROWS, DD);
  cvtf(e, e.sc4, 4, e.abf + DD, DD + 4, MROWS, 4);
  run_mlp3(e, nd, e.abf, DD + 4, DD + 4, DD, 1, e.sc1a, 1, ACT_SIGMOID);   // nprob
  cvtf(e, X, DD, e.abf, 2 * DD, MROWS, DD);
  cvtf(e, e.lmf, DD, e.abf + DD, 2 * DD, MROWS, DD);
  run_mlp3(e, rb, e.abf, 2 * DD, 2 * DD, DD, DD, e.t0, DD, ACT_NONE);      // ref
  blend_kernel<<<cdiv(MROWS * DD, 256), 256, 0, stream>>>(
      X, e.t0, e.sc1a, e.t1, MROWS, DD);                                   // den
  cvtf(e, X, DD, e.abf, DD + 1, MROWS, DD);
  cvtf(e, e.sc1a, 1, e.abf + DD, DD + 1, MROWS, 1);
  run_mlp3(e, cn, e.abf, DD + 1, DD + 1, 192, 1, e.sc1b, 1, ACT_SIGMOID);  // conf
  blend_kernel<<<cdiv(MROWS * DD, 256), 256, 0, stream>>>(
      X, e.t1, e.sc1b, e.t2, MROWS, DD);
  lnf(e, e.t2, DD, radm_norm, e.xB, DD, MROWS, DD, ACT_NONE);
  X = e.xB;

  // ================= head =================
  cvtf(e, X, DD, e.abf2, DD, MROWS, DD);
  gemmf(e, e.abf2, DD, head.w, 13, DD, head.b, out, 13, MROWS, ACT_NONE);
}
